// PointNetSetAbstraction_2482491097370
// MI455X (gfx1250) — matrix operand order repaired and checked by reading, not compiled, not run on hardware
//
#include <hip/hip_runtime.h>
#include <hip/hip_bf16.h>

typedef __attribute__((ext_vector_type(16))) _Float16 v16h;
typedef __attribute__((ext_vector_type(8)))  _Float16 v8h;
typedef __attribute__((ext_vector_type(8)))  float    v8f;

#define BS   4
#define NPT  2048
#define NC   512
#define CADD 64
#define CIN1 67
#define CP1  72   // layer1 channel pad (multiple of 8)
#define CO1  64
#define CO2  128
#define KB1  21   // ceil(9*72/32)
#define KB2  18   // 9*64/32
#define EPSBN 1e-5f

// ---------------------------------------------------------------------------
// 1) Furthest point sampling: one block per batch, 256 threads, 8 pts/thread.
// ---------------------------------------------------------------------------
__global__ __launch_bounds__(256)
void fps_kernel(const float* __restrict__ pts, const int* __restrict__ farInit,
                float* __restrict__ cent, float* __restrict__ outCent) {
    const int b = blockIdx.x, tid = threadIdx.x;
    __shared__ float sv[256];
    __shared__ int   si[256];
    __shared__ int   sFar;
    const float* pB = pts + (size_t)b * 3 * NPT;
    float dist[8];
#pragma unroll
    for (int i = 0; i < 8; ++i) dist[i] = 1e5f;
    if (tid == 0) sFar = farInit[b];
    __syncthreads();
    for (int it = 0; it < NC; ++it) {
        const int far = sFar;
        const float cx = pB[far], cy = pB[NPT + far], cz = pB[2 * NPT + far];
        if (tid < 3) {
            float cv = pB[tid * NPT + far];
            outCent[((size_t)b * 3 + tid) * NC + it] = cv;
            cent[((size_t)b * NC + it) * 3 + tid] = cv;
        }
        float best = -1.f; int bidx = 0;
#pragma unroll
        for (int i = 0; i < 8; ++i) {
            const int p = i * 256 + tid;
            float dx = pB[p] - cx, dy = pB[NPT + p] - cy, dz = pB[2 * NPT + p] - cz;
            float dd = dx * dx + dy * dy + dz * dz;
            if (dd < dist[i]) dist[i] = dd;
            if (dist[i] > best) { best = dist[i]; bidx = p; }
        }
        sv[tid] = best; si[tid] = bidx;
        __syncthreads();
        for (int s = 128; s > 0; s >>= 1) {
            if (tid < s) {
                float ov = sv[tid + s]; int oi = si[tid + s];
                if (ov > sv[tid] || (ov == sv[tid] && oi < si[tid])) { sv[tid] = ov; si[tid] = oi; }
            }
            __syncthreads();
        }
        if (tid == 0) sFar = si[0];
        __syncthreads();
    }
}

// ---------------------------------------------------------------------------
// 2) Stable rank of each centroid by distance to point j (j < num_nearby).
// ---------------------------------------------------------------------------
__global__ __launch_bounds__(512)
void rank_kernel(const float* __restrict__ pts, const float* __restrict__ cent,
                 int* __restrict__ gidx, int dnb) {
    const int b = blockIdx.x / dnb, j = blockIdx.x % dnb, i = threadIdx.x;
    __shared__ float sd[NC];
    const float px = pts[(size_t)b * 3 * NPT + j];
    const float py = pts[(size_t)b * 3 * NPT + NPT + j];
    const float pz = pts[(size_t)b * 3 * NPT + 2 * NPT + j];
    const float cx = cent[((size_t)b * NC + i) * 3 + 0];
    const float cy = cent[((size_t)b * NC + i) * 3 + 1];
    const float cz = cent[((size_t)b * NC + i) * 3 + 2];
    const float di = (px - cx) * (px - cx) + (py - cy) * (py - cy) + (pz - cz) * (pz - cz);
    sd[i] = di;
    __syncthreads();
    int rank = 0;
    for (int m = 0; m < NC; ++m) {
        float dm = sd[m];
        rank += (dm < di) || (dm == di && m < i);
    }
    gidx[((size_t)b * NC + rank) * dnb + j] = i;
}

// ---------------------------------------------------------------------------
// 3) Gather into f16 activation tensor act0[b][c][j][k]
// ---------------------------------------------------------------------------
__global__ __launch_bounds__(256)
void gather_kernel(const float* __restrict__ pts, const float* __restrict__ fv,
                   const int* __restrict__ gidx, int d, _Float16* __restrict__ act0,
                   int total) {
    int x = blockIdx.x * 256 + threadIdx.x;
    if (x >= total) return;
    const int k = x % NC;
    const int j = (x / NC) % d;
    const int c = (x / (NC * d)) % CIN1;
    const int b = x / (NC * d * CIN1);
    const int g = gidx[((size_t)b * NC + k) * d + j];
    float v = (c < 3) ? pts[((size_t)b * 3 + c) * NPT + g]
                      : fv[((size_t)b * CADD + (c - 3)) * NPT + g];
    act0[x] = (_Float16)v;
}

// ---------------------------------------------------------------------------
// 4) Pack conv weights into WMMA A-fragment layout, tap-major K mapping:
//    K = tap*Cpad + ci.  Lane L<16 holds row M=L, K = {0..7, 16..23};
//    lane L>=16 holds row M=L-16, K = {8..15, 24..31}.
// ---------------------------------------------------------------------------
__global__ __launch_bounds__(256)
void pack_weights(const float* __restrict__ W, int Cin, int Cpad, int KB,
                  _Float16* __restrict__ pA, int total) {
    int x = blockIdx.x * 256 + threadIdx.x;
    if (x >= total) return;
    const int t    = x & 15;
    const int lane = (x >> 4) & 31;
    const int rest = x >> 9;
    const int kb   = rest % KB;
    const int mt   = rest / KB;
    const int M    = mt * 16 + (lane & 15);
    const int base = (lane < 16) ? 0 : 8;
    const int kl   = (t < 8) ? (base + t) : (16 + base + (t - 8));
    const int K    = kb * 32 + kl;
    const int tap  = K / Cpad;
    const int ci   = K - tap * Cpad;
    float v = (tap < 9 && ci < Cin) ? W[((size_t)M * Cin + ci) * 9 + tap] : 0.f;
    pA[x] = (_Float16)v;
}

// compile-time LDS offset of im2col element e (tap-major K) in the
// channel-minor tile [10 rows][18 cols][Cpad]
__device__ constexpr int boffc(int e, int Cpad) {
    const int tap = e / Cpad;
    const int ci  = e - tap * Cpad;
    if (tap >= 9) return 0;     // K padding: A is zero there, value irrelevant
    const int dy = tap / 3;
    const int dx = tap - dy * 3;
    return (dy * 18 + dx) * Cpad + ci;
}

// ---------------------------------------------------------------------------
// 5) Implicit-GEMM 3x3 SAME conv via v_wmma_f32_16x16x32_f16.
//    kb outer / rows inner; B fragment = 2 contiguous ds_load_b128 per
//    lane-half (channel-minor tile + tap-major K => contiguous 8-chunks).
// ---------------------------------------------------------------------------
template <int TCIN, int TCPAD, int TCOUT, int TKB, int TD, int TBN>
__global__ __launch_bounds__(256)
void conv_wmma(const _Float16* __restrict__ actIn, const _Float16* __restrict__ pA,
               const float* __restrict__ bias,
               const float* __restrict__ bnSum, const float* __restrict__ bnSq,
               const float* __restrict__ gamma, const float* __restrict__ beta,
               _Float16* __restrict__ yOut) {
    constexpr int DH = TD / 8;
    constexpr float INVC = 1.f / (float)(BS * TD * NC);
    const int tid = threadIdx.x;
    const int wt  = blockIdx.x % 32;
    const int tmp = blockIdx.x / 32;
    const int h8  = tmp % DH;
    const int b   = tmp / DH;
    const int h0  = h8 * 8;

    __shared__ __align__(16) _Float16 tile[180 * TCPAD]; // [10 rows][18 cols][Cpad]
    __shared__ float sScale[TCIN];
    __shared__ float sShift[TCIN];

    if (tid < TCIN) {
        float s = 1.f, sh = 0.f;
        if (TBN) {
            float mu  = bnSum[tid] * INVC;
            float var = bnSq[tid] * INVC - mu * mu;
            s  = gamma[tid] * rsqrtf(var + EPSBN);
            sh = beta[tid] - mu * s;
        }
        sScale[tid] = s; sShift[tid] = sh;
    }
    __syncthreads();

    // stage input tile channel-minor (SAME padding -> zeros; BN+ReLU first)
    constexpr int STAGE = 180 * TCPAD;
    for (int x = tid; x < STAGE; x += 256) {
        const int c   = x % TCPAD;
        const int rc  = x / TCPAD;
        const int r   = rc / 18;
        const int col = rc - r * 18;
        const int hh  = h0 - 1 + r;
        const int ww  = wt * 16 - 1 + col;
        float f = 0.f;
        if (c < TCIN && hh >= 0 && hh < TD && ww >= 0 && ww < NC) {
            f = (float)actIn[(((size_t)b * TCIN + c) * TD + hh) * NC + ww];
            if (TBN) {
                f = sScale[c] * f + sShift[c];
                f = f > 0.f ? f : 0.f;
            }
        }
        tile[x] = (_Float16)f;
    }
    __syncthreads();

    const int lane = tid & 31;
    const int wave = tid >> 5;
    constexpr int MT = TCOUT >> 4;   // 4 or 8 output-channel tiles
    constexpr int NG = 8 / MT;       // 2 or 1 row groups
    constexpr int RR = MT;           // rows per wave (4 or 8)
    const int mt  = wave % MT;
    const int wg  = wave / MT;
    const int col = lane & 15;
    const int mb  = (lane < 16) ? 0 : 8;   // C/D row base for this lane half

    // per-row LDS base (element units): (hl*18 + col) * Cpad
    int rb[RR];
#pragma unroll
    for (int r = 0; r < RR; ++r) rb[r] = ((wg + r * NG) * 18 + col) * TCPAD;

    v8f acc[RR];
#pragma unroll
    for (int r = 0; r < RR; ++r) acc[r] = {};

#pragma unroll
    for (int kb = 0; kb < TKB; ++kb) {
        // A fragment: packed weights, loaded once per k-block, feeds RR WMMAs
        v16h a = *(const v16h*)(pA + (((size_t)mt * TKB + kb) * 32 + lane) * 16);
        if (kb + 1 < TKB)
            __builtin_prefetch(pA + (((size_t)mt * TKB + kb + 1) * 32 + lane) * 16, 0, 0);
#pragma unroll
        for (int r = 0; r < RR; ++r) {
            v8h lo, hi;
            if (lane < 16) {
                lo = *(const v8h*)&tile[rb[r] + boffc(kb * 32 + 0, TCPAD)];
                hi = *(const v8h*)&tile[rb[r] + boffc(kb * 32 + 16, TCPAD)];
            } else {
                lo = *(const v8h*)&tile[rb[r] + boffc(kb * 32 + 8, TCPAD)];
                hi = *(const v8h*)&tile[rb[r] + boffc(kb * 32 + 24, TCPAD)];
            }
            v16h bf = __builtin_shufflevector(lo, hi,
                0, 1, 2, 3, 4, 5, 6, 7, 8, 9, 10, 11, 12, 13, 14, 15);
            acc[r] = __builtin_amdgcn_wmma_f32_16x16x32_f16(
                false, a, false, bf, (short)0, acc[r], false, false);
        }
    }

    // epilogue: bias add, f16 store.  C/D layout: lane<16 -> rows 0..7,
    // lane>=16 -> rows 8..15, N = lane&15.
#pragma unroll
    for (int r = 0; r < RR; ++r) {
        const int h    = h0 + wg + r * NG;
        const int wcol = wt * 16 + col;
#pragma unroll
        for (int t = 0; t < 8; ++t) {
            const int co = mt * 16 + mb + t;
            float v = acc[r][t] + bias[co];
            yOut[(((size_t)b * TCOUT + co) * TD + h) * NC + wcol] = (_Float16)v;
        }
    }
}

// ---------------------------------------------------------------------------
// 6) Deterministic per-channel BN statistics: one block per channel.
// ---------------------------------------------------------------------------
__global__ __launch_bounds__(256)
void stats_kernel(const _Float16* __restrict__ y, int Cout, int d,
                  float* __restrict__ outSum, float* __restrict__ outSq) {
    const int co = blockIdx.x, tid = threadIdx.x;
    const int per = d * NC;
    __shared__ float ss[256];
    __shared__ float sq[256];
    float s = 0.f, q = 0.f;
    for (int x = tid; x < BS * per; x += 256) {
        const int b   = x / per;
        const int rem = x - b * per;
        float v = (float)y[((size_t)b * Cout + co) * per + rem];
        s += v; q += v * v;
    }
    ss[tid] = s; sq[tid] = q;
    __syncthreads();
    for (int st = 128; st > 0; st >>= 1) {
        if (tid < st) { ss[tid] += ss[tid + st]; sq[tid] += sq[tid + st]; }
        __syncthreads();
    }
    if (tid == 0) { outSum[co] = ss[0]; outSq[co] = sq[0]; }
}

// ---------------------------------------------------------------------------
// 7) BN + ReLU + max over neighbor axis (H=d) -> total_feature slice
// ---------------------------------------------------------------------------
__global__ __launch_bounds__(256)
void bn_relu_maxpool(const _Float16* __restrict__ y, const float* __restrict__ bnSum,
                     const float* __restrict__ bnSq, const float* __restrict__ gamma,
                     const float* __restrict__ beta, float invCnt, int Cout, int d,
                     int coOff, float* __restrict__ outFeat) {
    int x = blockIdx.x * 256 + threadIdx.x;
    if (x >= BS * CO2 * NC) return;
    const int k  = x % NC;
    const int co = (x / NC) % Cout;
    const int b  = x / (NC * Cout);
    const float mu  = bnSum[co] * invCnt;
    const float var = bnSq[co] * invCnt - mu * mu;
    const float s   = gamma[co] * rsqrtf(var + EPSBN);
    const float sh  = beta[co] - mu * s;
    float m = 0.f;   // ReLU outputs are >= 0
    const _Float16* yp = y + (((size_t)b * Cout + co) * d) * NC + k;
    for (int j = 0; j < d; ++j) {
        float v = s * (float)yp[(size_t)j * NC] + sh;
        v = v > 0.f ? v : 0.f;
        m = fmaxf(m, v);
    }
    outFeat[((size_t)b * 256 + coOff + co) * NC + k] = m;
}

// ---------------------------------------------------------------------------
// host launcher
// ---------------------------------------------------------------------------
extern "C" void kernel_launch(void* const* d_in, const int* in_sizes, int n_in,
                              void* d_out, int out_size, void* d_ws, size_t ws_size,
                              hipStream_t stream) {
    const float* pts     = (const float*)d_in[0];
    const float* fv      = (const float*)d_in[1];
    const int*   farInit = (const int*)d_in[2];
    float* outF = (float*)d_out;

    // workspace carving (256B aligned)
    char* base = (char*)d_ws;
    size_t o = 0;
    auto carve = [&](size_t bytes) -> char* {
        o = (o + 255) & ~(size_t)255;
        char* p = base + o;
        o += bytes;
        return p;
    };
    float*     cent  = (float*)carve((size_t)BS * NC * 3 * 4);
    int*       gidx  = (int*)carve((size_t)BS * NC * 64 * 4);
    float*     sum1  = (float*)carve(64 * 4);
    float*     sq1   = (float*)carve(64 * 4);
    float*     sum2  = (float*)carve(128 * 4);
    float*     sq2   = (float*)carve(128 * 4);
    _Float16*  pA1   = (_Float16*)carve((size_t)(CO1 / 16) * KB1 * 512 * 2);
    _Float16*  pA2   = (_Float16*)carve((size_t)(CO2 / 16) * KB2 * 512 * 2);
    _Float16*  act0  = (_Float16*)carve((size_t)BS * CIN1 * 64 * NC * 2);
    _Float16*  y1    = (_Float16*)carve((size_t)BS * CO1 * 64 * NC * 2);
    _Float16*  y2    = (_Float16*)carve((size_t)BS * CO2 * 64 * NC * 2);

    fps_kernel<<<BS, 256, 0, stream>>>(pts, farInit, cent, outF);

    for (int br = 0; br < 2; ++br) {
        const int d = (br == 0) ? 32 : 64;
        const int bi = 3 + br * 8;
        const float* W1  = (const float*)d_in[bi + 0];
        const float* b1  = (const float*)d_in[bi + 1];
        const float* g1  = (const float*)d_in[bi + 2];
        const float* be1 = (const float*)d_in[bi + 3];
        const float* W2  = (const float*)d_in[bi + 4];
        const float* b2  = (const float*)d_in[bi + 5];
        const float* g2  = (const float*)d_in[bi + 6];
        const float* be2 = (const float*)d_in[bi + 7];
        const float invCnt = 1.f / (float)(BS * d * NC);

        rank_kernel<<<BS * d, 512, 0, stream>>>(pts, cent, gidx, d);

        const int gtot = BS * CIN1 * d * NC;
        gather_kernel<<<(gtot + 255) / 256, 256, 0, stream>>>(pts, fv, gidx, d, act0, gtot);

        const int pt1 = (CO1 / 16) * KB1 * 512;
        pack_weights<<<(pt1 + 255) / 256, 256, 0, stream>>>(W1, CIN1, CP1, KB1, pA1, pt1);
        const int pt2 = (CO2 / 16) * KB2 * 512;
        pack_weights<<<(pt2 + 255) / 256, 256, 0, stream>>>(W2, CO1, CO1, KB2, pA2, pt2);

        const int cgrid = BS * (d / 8) * 32;
        if (br == 0) {
            conv_wmma<CIN1, CP1, CO1, KB1, 32, 0><<<cgrid, 256, 0, stream>>>(
                act0, pA1, b1, sum1, sq1, g1, be1, y1);
            stats_kernel<<<CO1, 256, 0, stream>>>(y1, CO1, d, sum1, sq1);
            conv_wmma<CO1, CO1, CO2, KB2, 32, 1><<<cgrid, 256, 0, stream>>>(
                y1, pA2, b2, sum1, sq1, g1, be1, y2);
        } else {
            conv_wmma<CIN1, CP1, CO1, KB1, 64, 0><<<cgrid, 256, 0, stream>>>(
                act0, pA1, b1, sum1, sq1, g1, be1, y1);
            stats_kernel<<<CO1, 256, 0, stream>>>(y1, CO1, d, sum1, sq1);
            conv_wmma<CO1, CO1, CO2, KB2, 64, 1><<<cgrid, 256, 0, stream>>>(
                y1, pA2, b2, sum1, sq1, g1, be1, y2);
        }
        stats_kernel<<<CO2, 256, 0, stream>>>(y2, CO2, d, sum2, sq2);

        const int mtot = BS * CO2 * NC;
        bn_relu_maxpool<<<(mtot + 255) / 256, 256, 0, stream>>>(
            y2, sum2, sq2, g2, be2, invCnt, CO2, d, br * 128, outF + (size_t)BS * 3 * NC);
    }
}
